// SimplifiedLlamaDecoderLayer_47141561041141
// MI455X (gfx1250) — matrix operand order repaired and checked by reading, not compiled, not run on hardware
//
#include <hip/hip_runtime.h>
#include <hip/hip_bf16.h>
#include <math.h>

// ---------------------------------------------------------------------------
// SimplifiedLlamaDecoderLayer for MI455X (gfx1250, wave32).
// - all matmuls: v_wmma_f32_16x16x32_bf16 (fp32 accumulate)
// - GEMM A-tiles staged with global_load_async_to_lds_b128 (ASYNCcnt)
// - bf16-resident operands everywhere (weights/activations converted once)
// ---------------------------------------------------------------------------

typedef __attribute__((ext_vector_type(16))) __bf16 v16bf;
typedef __attribute__((ext_vector_type(8)))  float  v8f;

union F8      { v8f v;   float f[8]; };
union BF16x16 { v16bf v; unsigned short s[16]; unsigned int u[8]; };

__device__ __forceinline__ unsigned short f2bf(float x) {
  union { __bf16 b; unsigned short u; } c; c.b = (__bf16)x; return c.u;
}
__device__ __forceinline__ float bf2f(unsigned short x) {
  union { __bf16 b; unsigned short u; } c; c.u = x; return (float)c.b;
}

// ---------------------------------------------------------------------------
// fp32 -> bf16 flat conversion (weights, once per launch)
// ---------------------------------------------------------------------------
__global__ __launch_bounds__(256) void cvt_bf16_kernel(
    const float* __restrict__ x, unsigned short* __restrict__ y, int total) {
  int idx = blockIdx.x * 256 + threadIdx.x;
  if (idx >= total) return;
  y[idx] = f2bf(x[idx]);
}

// ---------------------------------------------------------------------------
// RMSNorm: one row (C=1024) per 256-thread block; emits bf16.
// ---------------------------------------------------------------------------
__global__ __launch_bounds__(256) void rmsnorm_kernel(
    const float* __restrict__ x, const float* __restrict__ scale,
    unsigned short* __restrict__ out, int Cdim) {
  int row = blockIdx.x;
  const float* xr = x + (size_t)row * Cdim;
  unsigned short* yr = out + (size_t)row * Cdim;
  int t = threadIdx.x;
  float vals[4];
  float ss = 0.f;
  for (int i = 0; i < 4; ++i) { vals[i] = xr[t + i * 256]; ss += vals[i] * vals[i]; }
  for (int off = 16; off > 0; off >>= 1) ss += __shfl_xor(ss, off, 32);
  __shared__ float red[8];
  int wid = t >> 5, lane = t & 31;
  if (lane == 0) red[wid] = ss;
  __syncthreads();
  if (t == 0) { float s = 0.f; for (int i = 0; i < 8; ++i) s += red[i]; red[0] = s; }
  __syncthreads();
  float total = red[0];
  float rms = sqrtf(total / (float)Cdim);          // norm / sqrt(d)
  float inv = 1.f / (rms + 1e-8f);
  for (int i = 0; i < 4; ++i) {
    int c = t + i * 256;
    yr[c] = f2bf(scale[c] * vals[i] * inv);
  }
}

// ---------------------------------------------------------------------------
// Rotary diff (cos=sin=1): y[d] = x[d] - x[(d-1) mod 64] per head (bf16->bf16)
// ---------------------------------------------------------------------------
__global__ __launch_bounds__(256) void rot_bf16_kernel(
    const unsigned short* __restrict__ x, unsigned short* __restrict__ y, int total) {
  int idx = blockIdx.x * 256 + threadIdx.x;
  if (idx >= total) return;
  int base = idx & ~63;
  int d = idx & 63;
  int src = base + ((d + 63) & 63);
  y[idx] = f2bf(bf2f(x[idx]) - bf2f(x[src]));
}

// ---------------------------------------------------------------------------
// GEMM: act(A[M,K]bf16 @ B[K,N]bf16 + bias) (+resid) -> fp32 and/or bf16 out.
// Block tile 128x64, BK=64 (2 WMMA K-steps), 8 waves, each 32x32 (2x2 tiles).
// A tile staged via global_load_async_to_lds_b128 (ASYNCcnt); B tile staged
// transposed manually. LDS rows padded to 68 shorts (136B: 8B-aligned rows for
// B128 async writes; 34-dword stride -> conflict-free fragment reads).
// All dims are multiples of the tiles (no edge handling).
// ---------------------------------------------------------------------------
#define BM 128
#define BN 64
#define BKK 64
#define APAD 68   // shorts per A row (64 + 4)
#define BPAD 68   // shorts per B col

__global__ __launch_bounds__(256) void gemm_bf16_wmma(
    const unsigned short* __restrict__ A, const unsigned short* __restrict__ Bw,
    const float* __restrict__ bias, const float* __restrict__ resid,
    float* __restrict__ outF, unsigned short* __restrict__ outB,
    int M, int K, int Nn, int act) {
  __shared__ __align__(16) unsigned short aS[BM * APAD];
  __shared__ __align__(16) unsigned short bS[BN * BPAD];
  const unsigned int* aU = (const unsigned int*)aS;
  const unsigned int* bU = (const unsigned int*)bS;
  // LDS byte offset of aS within the workgroup allocation (low 32 bits of
  // the generic pointer are the LDS offset on AMDGPU).
  unsigned aBase = (unsigned)(unsigned long long)(void*)aS;
  unsigned long long Aaddr = (unsigned long long)A;

  int m0 = blockIdx.y * BM, n0 = blockIdx.x * BN;
  int tid = threadIdx.x;
  int wave = tid >> 5, lane = tid & 31;
  int wm = wave >> 1, wn = wave & 1;          // 4x2 wave grid
  int rowBase = wm * 32, colBase = wn * 32;
  int lh = lane >> 4, ll = lane & 15;

  F8 acc[2][2];
  for (int a = 0; a < 2; ++a)
    for (int b = 0; b < 2; ++b)
      for (int i = 0; i < 8; ++i) acc[a][b].f[i] = 0.f;

  for (int k0 = 0; k0 < K; k0 += BKK) {
    __syncthreads();
    // ---- A tile 128x64 bf16 via async B128 copies: 1024 16B chunks ----
    for (int c = tid; c < BM * 8; c += 256) {
      int r = c >> 3, cc = c & 7;
      unsigned ldsOff = aBase + (unsigned)(r * (APAD * 2) + cc * 16);
      unsigned gOff = (unsigned)(((m0 + r) * K + k0) * 2 + cc * 16);
      asm volatile("global_load_async_to_lds_b128 %0, %1, %2"
                   :: "v"(ldsOff), "v"(gOff), "s"(Aaddr) : "memory");
    }
    // ---- B tile 64x64 bf16, stored transposed bS[col][k] ----
    for (int idx = tid; idx < BKK * BN; idx += 256) {
      int kk = idx >> 6, nn = idx & 63;
      bS[nn * BPAD + kk] = Bw[(size_t)(k0 + kk) * Nn + (n0 + nn)];
    }
    // prefetch next B tile rows into L2/L0
    if (k0 + BKK < K)
      __builtin_prefetch(&Bw[(size_t)(k0 + BKK + (tid >> 6)) * Nn + n0 + (tid & 63)], 0, 1);
    asm volatile("s_wait_asynccnt 0" ::: "memory");
    __syncthreads();

    BF16x16 af[2][2], bfrg[2][2];   // [tile][kstep]
    for (int rt = 0; rt < 2; ++rt) {
      int row = rowBase + rt * 16 + ll;
      int kb2 = lh * 8;
      for (int ks = 0; ks < 2; ++ks)
        for (int i = 0; i < 4; ++i) {
          af[rt][ks].u[i]     = aU[row * (APAD / 2) + ks * 16 + kb2 + i];
          af[rt][ks].u[4 + i] = aU[row * (APAD / 2) + ks * 16 + kb2 + 4 + i];
        }
    }
    for (int ct = 0; ct < 2; ++ct) {
      int col = colBase + ct * 16 + ll;
      int kb2 = lh * 8;
      for (int ks = 0; ks < 2; ++ks)
        for (int i = 0; i < 8; ++i)
          bfrg[ct][ks].u[i] = bU[col * (BPAD / 2) + ks * 16 + kb2 + i];
    }
    for (int ks = 0; ks < 2; ++ks)
      for (int rt = 0; rt < 2; ++rt)
        for (int ct = 0; ct < 2; ++ct)
          acc[rt][ct].v = __builtin_amdgcn_wmma_f32_16x16x32_bf16(
              false, af[rt][ks].v, false, bfrg[ct][ks].v, (short)0,
              acc[rt][ct].v, false, false);
  }

  // epilogue: C/D layout -> row = base + lh*8 + v, col = base + ll
  for (int rt = 0; rt < 2; ++rt) {
    for (int ct = 0; ct < 2; ++ct) {
      int col = n0 + colBase + ct * 16 + ll;
      for (int vI = 0; vI < 8; ++vI) {
        int row = m0 + rowBase + rt * 16 + lh * 8 + vI;
        float val = acc[rt][ct].f[vI] + bias[col];
        if (act == 1) val = 0.5f * val * (1.f + erff(val * 0.70710678118f)); // exact GELU
        if (resid) val += resid[(size_t)row * Nn + col];
        if (outF) outF[(size_t)row * Nn + col] = val;
        if (outB) outB[(size_t)row * Nn + col] = f2bf(val);
      }
    }
  }
}

// ---------------------------------------------------------------------------
// Flash attention, one wave per (b, h, 16-query tile). D=64, key tiles of 32.
// S = Q K^T (4 WMMAs), online softmax, O += P V (4 WMMAs). bf16 in, bf16 out.
// ---------------------------------------------------------------------------
__global__ __launch_bounds__(32) void attn_kernel(
    const unsigned short* __restrict__ qb, const unsigned short* __restrict__ kb,
    const unsigned short* __restrict__ vb, unsigned short* __restrict__ outp,
    int Bn, int Nn, int Hn, int Dn) {
  __shared__ unsigned short Ps[16 * 34];
  const unsigned int* PsU = (const unsigned int*)Ps;
  int b = blockIdx.z, h = blockIdx.y, n0 = blockIdx.x * 16;
  int lane = threadIdx.x;
  int lh = lane >> 4, ll = lane & 15;
  int Cc = Hn * Dn;
  const unsigned int* qU = (const unsigned int*)qb;
  const unsigned int* kU = (const unsigned int*)kb;

  // Q fragments for both K-steps (channels [0,32) and [32,64) of the head)
  BF16x16 qf[2];
  {
    int n = n0 + ll;
    size_t base2 = ((size_t)(b * Nn + n) * Cc + h * Dn) >> 1;  // dword index
    int kb2 = lh * 8;
    for (int ks = 0; ks < 2; ++ks)
      for (int i = 0; i < 4; ++i) {
        qf[ks].u[i]     = qU[base2 + ks * 16 + kb2 + i];
        qf[ks].u[4 + i] = qU[base2 + ks * 16 + kb2 + 4 + i];
      }
  }

  float mrow[8], lrow[8], alpha[8];
  F8 o[4];
  for (int i = 0; i < 8; ++i) { mrow[i] = -3.0e38f; lrow[i] = 0.f; }
  for (int t = 0; t < 4; ++t)
    for (int i = 0; i < 8; ++i) o[t].f[i] = 0.f;

  const float sscale = 0.125f;  // 1/sqrt(64)

  for (int j0 = 0; j0 < Nn; j0 += 32) {
    // ---- S tile (16 queries x 32 keys) ----
    F8 s[2];
    for (int ct = 0; ct < 2; ++ct) {
      for (int i = 0; i < 8; ++i) s[ct].f[i] = 0.f;
      int j = j0 + ct * 16 + ll;
      size_t base2 = ((size_t)(b * Nn + j) * Cc + h * Dn) >> 1;
      int kb2 = lh * 8;
      for (int ks = 0; ks < 2; ++ks) {
        BF16x16 kf;
        for (int i = 0; i < 8; ++i) kf.u[i] = kU[base2 + ks * 16 + kb2 + i];
        s[ct].v = __builtin_amdgcn_wmma_f32_16x16x32_bf16(
            false, qf[ks].v, false, kf.v, (short)0, s[ct].v, false, false);
      }
    }
    // ---- online softmax (row r lives in VGPR r&7 of lane-half r>>3) ----
    for (int vI = 0; vI < 8; ++vI) {
      float a0 = s[0].f[vI] * sscale, a1 = s[1].f[vI] * sscale;
      float mv = fmaxf(a0, a1);
      for (int off = 1; off < 16; off <<= 1) mv = fmaxf(mv, __shfl_xor(mv, off, 32));
      float mnew = fmaxf(mrow[vI], mv);
      alpha[vI] = __expf(mrow[vI] - mnew);
      mrow[vI] = mnew;
      float p0 = __expf(a0 - mnew), p1 = __expf(a1 - mnew);
      s[0].f[vI] = p0; s[1].f[vI] = p1;
      float sum = p0 + p1;
      for (int off = 1; off < 16; off <<= 1) sum += __shfl_xor(sum, off, 32);
      lrow[vI] = alpha[vI] * lrow[vI] + sum;
    }
    for (int t = 0; t < 4; ++t)
      for (int vI = 0; vI < 8; ++vI) o[t].f[vI] *= alpha[vI];

    // ---- re-layout P: C/D layout -> A-fragment layout via LDS ----
    for (int vI = 0; vI < 8; ++vI) {
      int r = lh * 8 + vI;
      Ps[r * 34 + ll]      = f2bf(s[0].f[vI]);
      Ps[r * 34 + 16 + ll] = f2bf(s[1].f[vI]);
    }
    __syncthreads();
    BF16x16 pf;
    {
      int kb2 = lh * 8;
      for (int i = 0; i < 4; ++i) {
        pf.u[i]     = PsU[ll * 17 + kb2 + i];
        pf.u[4 + i] = PsU[ll * 17 + kb2 + 4 + i];
      }
    }
    __syncthreads();

    // ---- O += P (16x32) x V (32x16 per d-tile) ----
    for (int t = 0; t < 4; ++t) {
      BF16x16 vf;
      int d = t * 16 + ll;
      int kbase = lh * 16;
      for (int i = 0; i < 8; ++i) {
        int j = j0 + kbase + 2 * i;
        vf.s[2 * i]     = vb[(size_t)(b * Nn + j)     * Cc + h * Dn + d];
        vf.s[2 * i + 1] = vb[(size_t)(b * Nn + j + 1) * Cc + h * Dn + d];
      }
      o[t].v = __builtin_amdgcn_wmma_f32_16x16x32_bf16(
          false, pf.v, false, vf.v, (short)0, o[t].v, false, false);
    }
  }

  // ---- normalize and write (bf16) ----
  for (int vI = 0; vI < 8; ++vI) {
    float inv = 1.f / lrow[vI];
    int row = n0 + lh * 8 + vI;
    for (int t = 0; t < 4; ++t) {
      int d = t * 16 + ll;
      outp[(size_t)(b * Nn + row) * Cc + h * Dn + d] = f2bf(o[t].f[vI] * inv);
    }
  }
}

// ---------------------------------------------------------------------------
// Host-side orchestration
// ---------------------------------------------------------------------------
extern "C" void kernel_launch(void* const* d_in, const int* in_sizes, int n_in,
                              void* d_out, int out_size, void* d_ws, size_t ws_size,
                              hipStream_t stream) {
  (void)in_sizes; (void)n_in; (void)out_size; (void)ws_size;
  const float* x      = (const float*)d_in[0];
  const float* wq     = (const float*)d_in[1];
  const float* bq     = (const float*)d_in[2];
  const float* wk     = (const float*)d_in[3];
  const float* bk     = (const float*)d_in[4];
  const float* wv     = (const float*)d_in[5];
  const float* bv     = (const float*)d_in[6];
  const float* wo     = (const float*)d_in[7];
  const float* bo     = (const float*)d_in[8];
  const float* scale1 = (const float*)d_in[9];
  const float* scale2 = (const float*)d_in[10];
  const float* w1     = (const float*)d_in[11];
  const float* b1     = (const float*)d_in[12];
  const float* w2     = (const float*)d_in[13];
  const float* b2     = (const float*)d_in[14];
  float* out = (float*)d_out;

  const int Bn = 2, Nn = 2048, Cn = 1024, Hn = 16, Dn = 64, FFn = 4096;
  const int Mrows = Bn * Nn;  // 4096

  // workspace layout (~128 MB), all bf16 unless noted:
  char* ws = (char*)d_ws;
  const size_t MB = 1024 * 1024;
  unsigned short* xnb  = (unsigned short*)(ws);             // 8 MB
  unsigned short* wqb  = (unsigned short*)(ws + 8  * MB);   // 2 MB
  unsigned short* wkb  = (unsigned short*)(ws + 10 * MB);   // 2 MB
  unsigned short* wvb  = (unsigned short*)(ws + 12 * MB);   // 2 MB
  unsigned short* wob  = (unsigned short*)(ws + 14 * MB);   // 2 MB
  unsigned short* w1b  = (unsigned short*)(ws + 16 * MB);   // 8 MB
  unsigned short* w2b  = (unsigned short*)(ws + 24 * MB);   // 8 MB
  unsigned short* qtmp = (unsigned short*)(ws + 32 * MB);   // 8 MB
  unsigned short* ktmp = (unsigned short*)(ws + 40 * MB);   // 8 MB
  unsigned short* vbuf = (unsigned short*)(ws + 48 * MB);   // 8 MB
  unsigned short* qbuf = (unsigned short*)(ws + 56 * MB);   // 8 MB
  unsigned short* kbuf = (unsigned short*)(ws + 64 * MB);   // 8 MB
  unsigned short* atnb = (unsigned short*)(ws + 72 * MB);   // 8 MB
  float*          x2   = (float*)(ws + 80 * MB);            // 16 MB fp32
  unsigned short* hb   = (unsigned short*)(ws + 96 * MB);   // 32 MB

  // 0) weights -> bf16 (once per launch)
  int wsz = Cn * Cn, fsz = Cn * FFn;
  cvt_bf16_kernel<<<wsz / 256, 256, 0, stream>>>(wq, wqb, wsz);
  cvt_bf16_kernel<<<wsz / 256, 256, 0, stream>>>(wk, wkb, wsz);
  cvt_bf16_kernel<<<wsz / 256, 256, 0, stream>>>(wv, wvb, wsz);
  cvt_bf16_kernel<<<wsz / 256, 256, 0, stream>>>(wo, wob, wsz);
  cvt_bf16_kernel<<<fsz / 256, 256, 0, stream>>>(w1, w1b, fsz);
  cvt_bf16_kernel<<<fsz / 256, 256, 0, stream>>>(w2, w2b, fsz);

  // 1) xnb = rms_norm(x, scale1) in bf16
  rmsnorm_kernel<<<Mrows, 256, 0, stream>>>(x, scale1, xnb, Cn);

  // 2) q/k/v projections (bf16 WMMA, bf16 out)
  dim3 g1(Cn / BN, Mrows / BM);
  gemm_bf16_wmma<<<g1, 256, 0, stream>>>(xnb, wqb, bq, nullptr, nullptr, qtmp, Mrows, Cn, Cn, 0);
  gemm_bf16_wmma<<<g1, 256, 0, stream>>>(xnb, wkb, bk, nullptr, nullptr, ktmp, Mrows, Cn, Cn, 0);
  gemm_bf16_wmma<<<g1, 256, 0, stream>>>(xnb, wvb, bv, nullptr, nullptr, vbuf, Mrows, Cn, Cn, 0);

  // 3) rotary-diff for q,k
  int tot = Mrows * Cn;
  rot_bf16_kernel<<<tot / 256, 256, 0, stream>>>(qtmp, qbuf, tot);
  rot_bf16_kernel<<<tot / 256, 256, 0, stream>>>(ktmp, kbuf, tot);

  // 4) attention (flash-style, WMMA) -> bf16
  dim3 ga(Nn / 16, Hn, Bn);
  attn_kernel<<<ga, 32, 0, stream>>>(qbuf, kbuf, vbuf, atnb, Bn, Nn, Hn, Dn);

  // 5) output projection + residual -> x2 (fp32)
  gemm_bf16_wmma<<<g1, 256, 0, stream>>>(atnb, wob, bo, x, x2, nullptr, Mrows, Cn, Cn, 0);

  // 6) second norm
  rmsnorm_kernel<<<Mrows, 256, 0, stream>>>(x2, scale2, xnb, Cn);

  // 7) FFN up + exact GELU -> bf16
  dim3 g2(FFn / BN, Mrows / BM);
  gemm_bf16_wmma<<<g2, 256, 0, stream>>>(xnb, w1b, b1, nullptr, nullptr, hb, Mrows, Cn, FFn, 1);

  // 8) FFN down + residual -> out (fp32)
  dim3 g3(Cn / BN, Mrows / BM);
  gemm_bf16_wmma<<<g3, 256, 0, stream>>>(hb, w2b, b2, x2, out, nullptr, Mrows, FFn, Cn, 0);
}
